// Head_65489661329661
// MI455X (gfx1250) — matrix operand order repaired and checked by reading, not compiled, not run on hardware
//
#include <hip/hip_runtime.h>

// ---------------------------------------------------------------------------
// Single-head attention (B=4, T=4096, C=1024, hs=64) for gfx1250 (MI455X).
// Phase 1: qkv = x @ W^T via v_wmma_f32_16x16x32_f16 -> Q,K f16 [B*T,64],
//          V transposed f16 [B,64,T].
// Phase 2: flash attention; K/V tiles staged into double-buffered LDS by the
//          Tensor Data Mover (tensor_load_to_lds + s_wait_tensorcnt), one
//          wave per 16-query tile, key tiles of 32, online softmax in f32,
//          P re-laid out through per-wave LDS.
// ---------------------------------------------------------------------------

typedef __attribute__((ext_vector_type(16))) _Float16     v16h;
typedef __attribute__((ext_vector_type(8)))  _Float16     v8h;
typedef __attribute__((ext_vector_type(8)))  float        v8f;
typedef __attribute__((ext_vector_type(2)))  _Float16     h2;
typedef __attribute__((ext_vector_type(4)))  unsigned int u32x4;
typedef __attribute__((ext_vector_type(8)))  int          i32x8;
typedef __attribute__((ext_vector_type(4)))  int          i32x4;

#define BATCH 4
#define SEQ   4096
#define CDIM  1024
#define HS    64
#define KROW  72   // K-tile LDS row stride in halves (36 DWORDs, padded)
#define VROW  40   // V-tile LDS row stride in halves (20 DWORDs, padded)

__device__ __forceinline__ v8f wmma_f16(const v16h& a, const v16h& b, v8f c) {
    return __builtin_amdgcn_wmma_f32_16x16x32_f16(false, a, false, b,
                                                  (short)0, c, false, false);
}

__device__ __forceinline__ void lds_fence() {
    asm volatile("s_wait_dscnt 0" ::: "memory");
}

// TDM 2D tile load: tensor [td1 rows x td0 elems] of 2-byte elements with row
// stride stride0 (elems); tile [tld1 rows x tld0 elems] starting at gaddr.
// LDS destination padded: +('pad_amount'+1) DWORDs every 2^(pad_interval+1)
// DWORDs.  Descriptor packing per CDNA5 ISA.
__device__ __forceinline__ void tdm_load_2d(unsigned lds_off, unsigned long long gaddr,
                                            unsigned td0, unsigned td1,
                                            unsigned tld0, unsigned tld1,
                                            unsigned long long stride0,
                                            unsigned pad_interval, unsigned pad_amount)
{
    u32x4 g0;
    g0[0] = 1u;                                   // count=1, user mode, no gather
    g0[1] = lds_off;                              // LDS byte address
    g0[2] = (unsigned)gaddr;                      // global_addr[31:0]
    g0[3] = (unsigned)((gaddr >> 32) & 0x01FFFFFFull) | (2u << 30); // [56:32] | type=2
    i32x8 g1;
    g1[0] = (int)((1u << 16) |                    // data_size = 1 (2 bytes)
                  (1u << 20) |                    // pad_enable
                  (pad_interval << 22) | (pad_amount << 25));
    g1[1] = (int)((td0 & 0xFFFFu) << 16);                          // tensor_dim0 lo
    g1[2] = (int)(((td0 >> 16) & 0xFFFFu) | ((td1 & 0xFFFFu) << 16));
    g1[3] = (int)(((td1 >> 16) & 0xFFFFu) | ((tld0 & 0xFFFFu) << 16));
    g1[4] = (int)(tld1 & 0xFFFFu);                // tile_dim1 (tile_dim2 = 0)
    g1[5] = (int)(unsigned)stride0;               // tensor_dim0_stride lo
    g1[6] = (int)((unsigned)(stride0 >> 32) & 0xFFFFu);
    g1[7] = 0;
    i32x4 g2; g2[0] = 1; g2[1] = 1; g2[2] = 0; g2[3] = 0; // dims 2/3 trivial
    i32x4 g3; g3[0] = 0; g3[1] = 0; g3[2] = 0; g3[3] = 0;
#if defined(__clang_major__) && __clang_major__ >= 23
    i32x8 gz = {};
    __builtin_amdgcn_tensor_load_to_lds(g0, g1, g2, g3, gz, 0);
#else
    __builtin_amdgcn_tensor_load_to_lds(g0, g1, g2, g3, 0);
#endif
}

// ---------------------------------------------------------------------------
// Phase 1: QKV projection GEMM.  M = B*T = 16384, N = 192, K = 1024.
// Block = 4 waves; each wave: one 16-row strip x 12 N-tiles, K-steps of 32.
// ---------------------------------------------------------------------------
__global__ __launch_bounds__(128)
void qkv_kernel(const float* __restrict__ x, const float* __restrict__ W,
                _Float16* __restrict__ Qh, _Float16* __restrict__ Kh,
                _Float16* __restrict__ Vt)
{
    const int lane = threadIdx.x & 31;
    const int wave = threadIdx.x >> 5;
    const int h    = lane >> 4;
    const int l15  = lane & 15;
    const int Mbase = blockIdx.x * 64 + wave * 16;

    v8f acc[12];
#pragma unroll
    for (int nt = 0; nt < 12; ++nt) acc[nt] = {};

    for (int c0 = 0; c0 < CDIM; c0 += 32) {
        // A operand: x tile 16x32 (A(m,k), m = l15), fp32 -> f16.
        v16h a;
        const float* xrow = x + (size_t)(Mbase + l15) * CDIM + c0;
#pragma unroll
        for (int j = 0; j < 8; ++j) {
            const int k = h * 16 + (j << 1);
            float2 f = *(const float2*)(xrow + k);
            a[2 * j]     = (_Float16)f.x;
            a[2 * j + 1] = (_Float16)f.y;
        }
        // B operands double-buffered to avoid WAR on in-flight WMMA B regs.
        auto loadB = [&](v16h& b, int nt) {
            const float* wrow = W + (size_t)(nt * 16 + l15) * CDIM + c0;
#pragma unroll
            for (int j = 0; j < 8; ++j) {
                float2 f = *(const float2*)(wrow + h * 16 + (j << 1));
                b[2 * j]     = (_Float16)f.x;
                b[2 * j + 1] = (_Float16)f.y;
            }
        };
        v16h bnext;
        loadB(bnext, 0);
#pragma unroll
        for (int nt = 0; nt < 12; ++nt) {
            v16h bcur = bnext;
            if (nt < 11) loadB(bnext, nt + 1);
            acc[nt] = wmma_f16(a, bcur, acc[nt]);
        }
    }

    // Epilogue: nt 0-3 -> Q, 4-7 -> K (b16, lane-coalesced), 8-11 -> V^T
    // (8 rows packed into one b128 per lane: consecutive t, fixed channel).
#pragma unroll
    for (int nt = 0; nt < 8; ++nt) {
        const int ch = (nt & 3) * 16 + l15;
        _Float16* dst = (nt < 4) ? Qh : Kh;
#pragma unroll
        for (int r = 0; r < 8; ++r) {
            const int row = Mbase + h * 8 + r;
            dst[(size_t)row * HS + ch] = (_Float16)acc[nt][r];
        }
    }
    {
        const int bb    = Mbase >> 12;
        const int tbase = (Mbase & (SEQ - 1)) + h * 8;
#pragma unroll
        for (int nt = 8; nt < 12; ++nt) {
            v8h pk;
#pragma unroll
            for (int r = 0; r < 8; ++r) pk[r] = (_Float16)acc[nt][r];
            const int ch = (nt - 8) * 16 + l15;
            *(v8h*)(Vt + ((size_t)bb * HS + ch) * SEQ + tbase) = pk;
        }
    }
}

// ---------------------------------------------------------------------------
// Phase 2: flash attention.  Block = 8 waves; wave owns 16 queries; key tiles
// of 32.  Wave 0 drives the TDM double-buffer for the block's shared K/V.
// ---------------------------------------------------------------------------
__global__ __launch_bounds__(256)
void attn_kernel(const _Float16* __restrict__ Qh, const _Float16* __restrict__ Kh,
                 const _Float16* __restrict__ Vt, const unsigned char* __restrict__ mask,
                 float* __restrict__ out)
{
    __shared__ __attribute__((aligned(16))) _Float16 kbuf[2][32 * KROW];
    __shared__ __attribute__((aligned(16))) _Float16 vbuf[2][64 * VROW];
    __shared__ __attribute__((aligned(16))) _Float16 pbuf[8 * 16 * 34];

    const int lane = threadIdx.x & 31;
    const int wave = threadIdx.x >> 5;
    const int h    = lane >> 4;
    const int l15  = lane & 15;
    const int row0 = (blockIdx.x * 8 + wave) * 16;  // global query row
    const int b    = row0 >> 12;                    // whole block same batch
    _Float16* P = pbuf + wave * 16 * 34;
    const float scale = 0.125f;                     // hs^-0.5

    // Q tile as two resident A operands (K-chunks of 32).
    v16h aq[2];
#pragma unroll
    for (int kc = 0; kc < 2; ++kc) {
#pragma unroll
        for (int j = 0; j < 8; ++j) {
            const int k = kc * 32 + h * 16 + (j << 1);
            h2 p = *(const h2*)(Qh + (size_t)(row0 + l15) * HS + k);
            aq[kc][2 * j] = p.x; aq[kc][2 * j + 1] = p.y;
        }
    }

    v8f o[4];
    float m[8], l[8];
#pragma unroll
    for (int nt = 0; nt < 4; ++nt) o[nt] = {};
#pragma unroll
    for (int r = 0; r < 8; ++r) { m[r] = -1e30f; l[r] = 0.0f; }

    const unsigned long long kg =
        (unsigned long long)(size_t)(Kh + (size_t)b * SEQ * HS);
    const unsigned long long vg =
        (unsigned long long)(size_t)(Vt + (size_t)b * HS * SEQ);
    const unsigned char* mbase = mask + (size_t)b * SEQ;

    // TDM issue for key tile kt2 into LDS buffer `buf` (wave 0 only).
    auto issue = [&](int kt2, int buf) {
        // K tile: 32 rows x 64 halves, tensor row stride 64; pad 32dw+4dw.
        tdm_load_2d((unsigned)(size_t)(void*)&kbuf[buf][0],
                    kg + (unsigned long long)kt2 * 32 * HS * 2,
                    HS, SEQ, HS, 32, HS, /*pad_interval=*/4, /*pad_amount=*/3);
        // V^T tile: 64 rows x 32 halves, tensor row stride 4096; pad 16dw+4dw.
        tdm_load_2d((unsigned)(size_t)(void*)&vbuf[buf][0],
                    vg + (unsigned long long)kt2 * 32 * 2,
                    SEQ, HS, 32, HS, SEQ, /*pad_interval=*/3, /*pad_amount=*/3);
    };
    if (wave == 0) issue(0, 0);

    for (int kt = 0; kt < SEQ / 32; ++kt) {
        const int cur = kt & 1;
        const int kb  = kt * 32;
        if (wave == 0) __builtin_amdgcn_s_wait_tensorcnt(0);
        __syncthreads();                       // tile kt landed for everyone
        asm volatile("" ::: "memory");
        if (wave == 0 && kt + 1 < SEQ / 32) issue(kt + 1, cur ^ 1);

        const _Float16* kk = &kbuf[cur][0];    // [32][KROW]
        const _Float16* vv = &vbuf[cur][0];    // [64][VROW]

        // ---- S = Q @ K^T : 2 N-tiles of 16 keys, K-reduce over hs=64 ----
        v8f s[2];
#pragma unroll
        for (int nt = 0; nt < 2; ++nt) {
            v8f sc = {};
#pragma unroll
            for (int kc = 0; kc < 2; ++kc) {
                v16h bk;
                const _Float16* krow = kk + (nt * 16 + l15) * KROW + kc * 32 + h * 16;
#pragma unroll
                for (int j = 0; j < 8; ++j) {
                    h2 p = *(const h2*)(krow + (j << 1));
                    bk[2 * j] = p.x; bk[2 * j + 1] = p.y;
                }
                sc = wmma_f16(aq[kc], bk, sc);
            }
            s[nt] = sc;
        }

        // ---- online softmax; per-row stats live replicated per 16-group ----
        const bool keep0 = mbase[kb + l15] != 0;
        const bool keep1 = mbase[kb + 16 + l15] != 0;
#pragma unroll
        for (int r = 0; r < 8; ++r) {
            float s0 = keep0 ? s[0][r] * scale : -1e30f;
            float s1 = keep1 ? s[1][r] * scale : -1e30f;
            float mx = fmaxf(s0, s1);
#pragma unroll
            for (int off = 8; off >= 1; off >>= 1)
                mx = fmaxf(mx, __shfl_xor(mx, off, 32));
            const float mnew  = fmaxf(m[r], mx);
            const float alpha = __expf(m[r] - mnew);
            const float p0 = __expf(s0 - mnew);
            const float p1 = __expf(s1 - mnew);
            float rs = p0 + p1;
#pragma unroll
            for (int off = 8; off >= 1; off >>= 1)
                rs += __shfl_xor(rs, off, 32);
            l[r] = l[r] * alpha + rs;
            m[r] = mnew;
#pragma unroll
            for (int nt = 0; nt < 4; ++nt) o[nt][r] *= alpha;
            P[(h * 8 + r) * 34 + l15]      = (_Float16)p0;
            P[(h * 8 + r) * 34 + 16 + l15] = (_Float16)p1;
        }
        lds_fence();   // per-wave P region; LDS in-order, block reordering

        // ---- reload P in A layout (16x32, K = key-in-tile) ----
        v16h ap;
#pragma unroll
        for (int j = 0; j < 8; ++j) {
            const int k = h * 16 + (j << 1);
            h2 p = *(const h2*)(P + l15 * 34 + k);
            ap[2 * j] = p.x; ap[2 * j + 1] = p.y;
        }

        // ---- O += P @ V : B(k,n=c) from staged V^T tile ----
#pragma unroll
        for (int nt = 0; nt < 4; ++nt) {
            v16h bv;
            const _Float16* vrow = vv + (nt * 16 + l15) * VROW + h * 16;
#pragma unroll
            for (int j = 0; j < 8; ++j) {
                h2 p = *(const h2*)(vrow + (j << 1));
                bv[2 * j] = p.x; bv[2 * j + 1] = p.y;
            }
            o[nt] = wmma_f16(ap, bv, o[nt]);
        }
    }

    // ---- normalize and write fp32 output [B*T, 64] ----
#pragma unroll
    for (int r = 0; r < 8; ++r) {
        const float inv = 1.0f / l[r];
#pragma unroll
        for (int nt = 0; nt < 4; ++nt) {
            out[(size_t)(row0 + h * 8 + r) * HS + nt * 16 + l15] = o[nt][r] * inv;
        }
    }
}

// ---------------------------------------------------------------------------
extern "C" void kernel_launch(void* const* d_in, const int* in_sizes, int n_in,
                              void* d_out, int out_size, void* d_ws, size_t ws_size,
                              hipStream_t stream) {
    const float*         x    = (const float*)d_in[0];
    const unsigned char* mask = (const unsigned char*)d_in[1]; // jnp bool = 1 byte
    const float*         W    = (const float*)d_in[2];
    float*               out  = (float*)d_out;

    _Float16* Qh = (_Float16*)d_ws;                    // 2 MB
    _Float16* Kh = Qh + (size_t)BATCH * SEQ * HS;      // 2 MB
    _Float16* Vt = Kh + (size_t)BATCH * SEQ * HS;      // 2 MB

    qkv_kernel<<<dim3(256), dim3(128), 0, stream>>>(x, W, Qh, Kh, Vt);
    attn_kernel<<<dim3(128), dim3(256), 0, stream>>>(Qh, Kh, Vt, mask, out);
}
